// MossTTSLocalTransformer_66348654788746
// MI455X (gfx1250) — matrix operand order repaired and checked by reading, not compiled, not run on hardware
//
#include <hip/hip_runtime.h>
#include <hip/hip_bf16.h>
#include <math.h>

// MossTTS local transformer decode on MI455X (gfx1250).
// seq_len==1 => softmax over one key == 1 => attention output = repeat(v)@wo;
// q/k are dead. Head-repeat is pre-folded into wo (512-K GEMM). Weights are
// converted once to bf16 [N,K] so the hot set (~150MB) stays L2-resident
// across the 16 sequential codebook steps; all GEMMs run on
// v_wmma_f32_16x16x32_bf16. All GEMM N dims are padded to multiples of 256
// (lm head: 1027 -> 1280 with zeroed pad rows) so the inner loop has zero
// divergence / exec manipulation.

typedef __bf16 bf16_t;
typedef __attribute__((ext_vector_type(16))) __bf16 v16bf;
typedef __attribute__((ext_vector_type(8)))  __bf16 v8bf;
typedef __attribute__((ext_vector_type(8)))  float  v8f;

// ---------------- GEMM: C[M,N] = A[M,K](bf16) * Bt[N,K](bf16, pre-transposed)
// Requirements: M % 32 == 0, N % 256 == 0, K % 32 == 0 (padding guarantees).
// Block = 256 thr = 8 waves: 2 over M x 4 over N -> 32x256 block tile.
// Wave = 16(M) x 64(N): 4 accumulators share one A fragment per K=32 step.
__global__ __launch_bounds__(256) void gemm_bf16_wmma(
    const bf16_t* __restrict__ A, const bf16_t* __restrict__ Bt,
    const float* __restrict__ addsrc, float* __restrict__ Cf,
    bf16_t* __restrict__ Cb, int M, int N, int K)
{
  const int lane = threadIdx.x & 31;
  const int wave = threadIdx.x >> 5;
  const int hx = lane >> 4;   // wave half (0/1)
  const int lx = lane & 15;

  const int m0 = blockIdx.y * 32 + (wave & 1) * 16;
  const int n0 = blockIdx.x * 256 + (wave >> 1) * 64;

  // A: lane half hx holds 16 contiguous K = {16hx..16hx+15} (matches B)
  const bf16_t* arow = A + (size_t)(m0 + lx) * K + hx * 16;
  // B: lane holds 16 contiguous K at fixed N (Bt is [N,K])
  const bf16_t* bp0 = Bt + (size_t)(n0 + lx +  0) * K + hx * 16;
  const bf16_t* bp1 = Bt + (size_t)(n0 + lx + 16) * K + hx * 16;
  const bf16_t* bp2 = Bt + (size_t)(n0 + lx + 32) * K + hx * 16;
  const bf16_t* bp3 = Bt + (size_t)(n0 + lx + 48) * K + hx * 16;

  v8f acc0 = {}, acc1 = {}, acc2 = {}, acc3 = {};

#pragma unroll 2
  for (int k0 = 0; k0 < K; k0 += 32) {
    v8bf alo = *(const v8bf*)(arow + k0);
    v8bf ahi = *(const v8bf*)(arow + k0 + 8);
    v16bf a;
#pragma unroll
    for (int j = 0; j < 8; ++j) { a[j] = alo[j]; a[8 + j] = ahi[j]; }

    v16bf b0 = *(const v16bf*)(bp0 + k0);
    v16bf b1 = *(const v16bf*)(bp1 + k0);
    v16bf b2 = *(const v16bf*)(bp2 + k0);
    v16bf b3 = *(const v16bf*)(bp3 + k0);

    acc0 = __builtin_amdgcn_wmma_f32_16x16x32_bf16(false, a, false, b0,
                                                   (short)0, acc0, false, false);
    acc1 = __builtin_amdgcn_wmma_f32_16x16x32_bf16(false, a, false, b1,
                                                   (short)0, acc1, false, false);
    acc2 = __builtin_amdgcn_wmma_f32_16x16x32_bf16(false, a, false, b2,
                                                   (short)0, acc2, false, false);
    acc3 = __builtin_amdgcn_wmma_f32_16x16x32_bf16(false, a, false, b3,
                                                   (short)0, acc3, false, false);
  }

  // C/D layout: VGPR r -> row m0 + 8*hx + r, col = n0 + 16*t + lx
  v8f accs[4] = {acc0, acc1, acc2, acc3};
#pragma unroll
  for (int t = 0; t < 4; ++t) {
    int col = n0 + t * 16 + lx;
#pragma unroll
    for (int rr = 0; rr < 8; ++rr) {
      size_t off = (size_t)(m0 + hx * 8 + rr) * N + col;
      float v = accs[t][rr];
      if (addsrc) v += addsrc[off];
      if (Cf) Cf[off] = v;
      if (Cb) Cb[off] = (bf16_t)v;
    }
  }
}

// ---------------- RMSNorm: out_bf16[b,:] = h[b,:] * rsqrt(mean(h^2)+eps) * w
__global__ __launch_bounds__(256) void rms_kernel(
    const float* __restrict__ h, const float* __restrict__ w,
    bf16_t* __restrict__ out, int Hd)
{
  int b = blockIdx.x, t = threadIdx.x;
  __shared__ float red[256];
  float s = 0.f;
  for (int j = t; j < Hd; j += 256) { float v = h[(size_t)b * Hd + j]; s += v * v; }
  red[t] = s; __syncthreads();
  for (int off = 128; off > 0; off >>= 1) {
    if (t < off) red[t] += red[t + off];
    __syncthreads();
  }
  float inv = rsqrtf(red[0] / (float)Hd + 1e-6f);
  for (int j = t; j < Hd; j += 256)
    out[(size_t)b * Hd + j] = (bf16_t)(h[(size_t)b * Hd + j] * inv * w[j]);
}

// ---------------- SwiGLU over fused gate|up buffer gu[256][8192]
__global__ __launch_bounds__(256) void swiglu_kernel(
    const float* __restrict__ gu, bf16_t* __restrict__ out, int Im)
{
  long i = (long)blockIdx.x * 256 + threadIdx.x;   // over 256*Im
  long b = i / Im;
  int  n = (int)(i - b * Im);
  float g = gu[b * 2 * Im + n];
  float u = gu[b * 2 * Im + Im + n];
  out[i] = (bf16_t)((g / (1.f + __expf(-g))) * u);
}

// ---------------- fp32 [K,N] -> bf16 [N,K] (one-time weight prep)
__global__ __launch_bounds__(256) void tconv_kernel(
    const float* __restrict__ in, bf16_t* __restrict__ out, int K, int N)
{
  long idx = (long)blockIdx.x * 256 + threadIdx.x;
  if (idx >= (long)K * N) return;
  long k = idx / N;
  int n = (int)(idx - k * N);
  out[(size_t)n * K + k] = (bf16_t)in[idx];
}

// ---------------- lm heads fp32 [C,V,H] -> bf16 [C,Vp,H], zero pad rows
__global__ __launch_bounds__(256) void convpad_kernel(
    const float* __restrict__ in, bf16_t* __restrict__ out,
    int Vv, int Vp, int Hd)
{
  long i = (long)blockIdx.x * 256 + threadIdx.x;   // over Cc*Vp*Hd
  long rs = i / Hd;
  int  k  = (int)(i - rs * Hd);
  long s  = rs / Vp;
  int  r  = (int)(rs - s * Vp);
  out[i] = (r < Vv) ? (bf16_t)in[((size_t)s * Vv + r) * Hd + k] : (bf16_t)0.f;
}

// ---------------- fold head-repeat into wo: out[n*512+k] = wo[i1,n]+wo[i2,n]
__global__ __launch_bounds__(256) void fold_wo_kernel(
    const float* __restrict__ wo, bf16_t* __restrict__ out)
{
  int idx = blockIdx.x * 256 + threadIdx.x;      // over 1024*512
  int n = idx >> 9, k = idx & 511;
  int h2 = k >> 6, d = k & 63;
  float a = wo[(size_t)(128 * h2 + d) * 1024 + n];
  float b = wo[(size_t)(128 * h2 + 64 + d) * 1024 + n];
  out[(size_t)n * 512 + k] = (bf16_t)(a + b);
}

// ---------------- repetition penalty + (deterministic) argmax sample
__global__ __launch_bounds__(256) void sample_kernel(
    const float* __restrict__ logits, int ldl, const int* __restrict__ hist,
    const int* __restrict__ gen_step_p, int step_i,
    int* __restrict__ tok, int* __restrict__ out_tokens)
{
  const int Vv = 1027, Cc = 16, Lh = 200, W = 50;
  int b = blockIdx.x, t = threadIdx.x;
  int gs = gen_step_p[0];
  int w0 = gs - W; if (w0 < 0) w0 = 0;
  int wlen = gs - w0; if (wlen > W) wlen = W;

  __shared__ float pen[50];
  __shared__ int pentok[50];
  if (t < wlen) {
    int id = hist[(size_t)b * Lh * Cc + (size_t)(w0 + t) * Cc + step_i];
    float cur = logits[(size_t)b * ldl + id];
    pen[t] = (cur < 0.f) ? cur * 1.1f : cur / 1.1f;
    pentok[t] = id;
  }
  __syncthreads();

  float best = -INFINITY; int bi = 0;
  for (int v = t; v < Vv; v += 256) {
    float val = logits[(size_t)b * ldl + v];
    for (int j = 0; j < wlen; ++j)
      if (pentok[j] == v) { val = pen[j]; break; }
    if (val > best || (val == best && v < bi)) { best = val; bi = v; }
  }

  __shared__ float sv[256]; __shared__ int si[256];
  sv[t] = best; si[t] = bi; __syncthreads();
  for (int off = 128; off > 0; off >>= 1) {
    if (t < off) {
      if (sv[t + off] > sv[t] || (sv[t + off] == sv[t] && si[t + off] < si[t])) {
        sv[t] = sv[t + off]; si[t] = si[t + off];
      }
    }
    __syncthreads();
  }
  if (t == 0) { tok[b] = si[0]; out_tokens[b * Cc + step_i] = si[0]; }
}

// ---------------- embedding gather for next step hidden
__global__ __launch_bounds__(256) void embed_kernel(
    const float* __restrict__ tables, const int* __restrict__ tok,
    float* __restrict__ h, int next_i)
{
  const int Vv = 1027, Hd = 1024;
  int b = blockIdx.x;
  const float* src = tables + ((size_t)next_i * Vv + tok[b]) * Hd;
  for (int j = threadIdx.x; j < Hd; j += 256)
    h[(size_t)b * Hd + j] = src[j];
}

// ================= host orchestration =================
static inline void launch_gemm(const bf16_t* A, const bf16_t* Bt,
                               const float* add, float* Cf, bf16_t* Cb,
                               int M, int N, int K, hipStream_t s)
{
  dim3 grid(N / 256, M / 32);
  gemm_bf16_wmma<<<grid, 256, 0, s>>>(A, Bt, add, Cf, Cb, M, N, K);
}

extern "C" void kernel_launch(void* const* d_in, const int* in_sizes, int n_in,
                              void* d_out, int out_size, void* d_ws, size_t ws_size,
                              hipStream_t stream)
{
  (void)in_sizes; (void)n_in; (void)out_size; (void)ws_size;
  const int Bsz = 256, Hd = 1024, Dv = 512, Im = 4096, Vv = 1027, Vp = 1280,
            Cc = 16, Ll = 4;

  const float* backbone  = (const float*)d_in[0];
  const int*   hist      = (const int*)d_in[1];
  const int*   gen_step  = (const int*)d_in[2];
  const float* embeds    = (const float*)d_in[3];
  const float* lm_heads  = (const float*)d_in[4];
  const float* w_in_norm = (const float*)d_in[5];
  // d_in[6] w_q, d_in[7] w_k, d_in[9] q_norm, d_in[10] k_norm: dead (seq_len==1)
  const float* w_v       = (const float*)d_in[8];
  const float* w_o       = (const float*)d_in[11];
  const float* w_post    = (const float*)d_in[12];
  const float* w_gate    = (const float*)d_in[13];
  const float* w_up      = (const float*)d_in[14];
  const float* w_down    = (const float*)d_in[15];
  const float* fin_norm  = (const float*)d_in[16];

  char* p = (char*)d_ws;
  auto carve = [&](size_t bytes) {
    void* r = (void*)p;
    p += (bytes + 255) & ~(size_t)255;
    return r;
  };
  bf16_t* wv_t  = (bf16_t*)carve((size_t)Ll * Dv * Hd * 2);      // [512,1024]/L
  bf16_t* wo_f  = (bf16_t*)carve((size_t)Ll * Hd * Dv * 2);      // [1024,512]/L
  bf16_t* wgu_t = (bf16_t*)carve((size_t)Ll * 2 * Im * Hd * 2);  // [8192,1024]/L
  bf16_t* wd_t  = (bf16_t*)carve((size_t)Ll * Hd * Im * 2);      // [1024,4096]/L
  bf16_t* lm_b  = (bf16_t*)carve((size_t)Cc * Vp * Hd * 2);      // [1280,1024]/step
  float*  h     = (float*) carve((size_t)Bsz * Hd * 4);
  bf16_t* xb    = (bf16_t*)carve((size_t)Bsz * Hd * 2);
  bf16_t* vb    = (bf16_t*)carve((size_t)Bsz * Dv * 2);
  float*  gu    = (float*) carve((size_t)Bsz * 2 * Im * 4);      // gate|up fused
  bf16_t* mb    = (bf16_t*)carve((size_t)Bsz * Im * 2);
  float*  lg    = (float*) carve((size_t)Bsz * Vp * 4);
  int*    tok   = (int*)   carve((size_t)Bsz * 4);

  // ---- one-time weight prep: bf16 + [N,K] transpose (~150MB, L2-resident)
  for (int l = 0; l < Ll; ++l) {
    long nvw = (long)Hd * Dv;
    tconv_kernel<<<(int)((nvw + 255) / 256), 256, 0, stream>>>(
        w_v + (size_t)l * Hd * Dv, wv_t + (size_t)l * Dv * Hd, Hd, Dv);
    long ngw = (long)Hd * Im;
    // gate -> rows [0,4096), up -> rows [4096,8192) of fused B
    tconv_kernel<<<(int)((ngw + 255) / 256), 256, 0, stream>>>(
        w_gate + (size_t)l * Hd * Im, wgu_t + (size_t)l * 2 * Im * Hd, Hd, Im);
    tconv_kernel<<<(int)((ngw + 255) / 256), 256, 0, stream>>>(
        w_up + (size_t)l * Hd * Im,
        wgu_t + (size_t)l * 2 * Im * Hd + (size_t)Im * Hd, Hd, Im);
    long ndw = (long)Im * Hd;
    tconv_kernel<<<(int)((ndw + 255) / 256), 256, 0, stream>>>(
        w_down + (size_t)l * Im * Hd, wd_t + (size_t)l * Hd * Im, Im, Hd);
    fold_wo_kernel<<<(Hd * Dv) / 256, 256, 0, stream>>>(
        w_o + (size_t)l * Hd * Hd, wo_f + (size_t)l * Hd * Dv);
  }
  long nlm = (long)Cc * Vp * Hd;
  convpad_kernel<<<(int)((nlm + 255) / 256), 256, 0, stream>>>(
      lm_heads, lm_b, Vv, Vp, Hd);

  hipMemcpyAsync(h, backbone, (size_t)Bsz * Hd * 4, hipMemcpyDeviceToDevice,
                 stream);

  // ---- 16 sequential codebook steps
  for (int i = 0; i < Cc; ++i) {
    for (int l = 0; l < Ll; ++l) {
      rms_kernel<<<Bsz, 256, 0, stream>>>(h, w_in_norm + (size_t)l * Hd, xb, Hd);
      // v = x @ wv (attention collapses to repeat(v) @ wo at seq_len==1)
      launch_gemm(xb, wv_t + (size_t)l * Dv * Hd, nullptr, nullptr, vb,
                  Bsz, Dv, Hd, stream);
      // h += v @ wo_folded (in-place residual epilogue)
      launch_gemm(vb, wo_f + (size_t)l * Hd * Dv, h, h, nullptr,
                  Bsz, Hd, Dv, stream);
      rms_kernel<<<Bsz, 256, 0, stream>>>(h, w_post + (size_t)l * Hd, xb, Hd);
      // fused [gate|up] GEMM: [256,1024] x [1024,8192]
      launch_gemm(xb, wgu_t + (size_t)l * 2 * Im * Hd, nullptr, gu, nullptr,
                  Bsz, 2 * Im, Hd, stream);
      swiglu_kernel<<<(Bsz * Im) / 256, 256, 0, stream>>>(gu, mb, Im);
      // h += m @ wd
      launch_gemm(mb, wd_t + (size_t)l * Hd * Im, h, h, nullptr,
                  Bsz, Hd, Im, stream);
    }
    rms_kernel<<<Bsz, 256, 0, stream>>>(h, fin_norm, xb, Hd);
    launch_gemm(xb, lm_b + (size_t)i * Vp * Hd, nullptr, lg, nullptr,
                Bsz, Vp, Hd, stream);
    sample_kernel<<<Bsz, 256, 0, stream>>>(lg, Vp, hist, gen_step, i, tok,
                                           (int*)d_out);
    if (i < Cc - 1)
      embed_kernel<<<Bsz, 256, 0, stream>>>(embeds, tok, h, i + 1);
  }
}
